// DeformableConv2d_4252017623066
// MI455X (gfx1250) — compile-verified, matrix operand order checked by reading, not run on hardware
//
#include <hip/hip_runtime.h>

typedef _Float16 v16h __attribute__((ext_vector_type(16)));
typedef float    v8f  __attribute__((ext_vector_type(8)));

#define Bn    4
#define Cin   128
#define Hh    128
#define Ww    128
#define Oout  256
#define KK    9
#define Ktot  (Cin*KK)        // 1152
#define NPIX  16
#define KBLK  32              // K per WMMA instruction
#define NKB   (Ktot/KBLK)     // 36 k-blocks total
#define CCH   32              // channels per LDS chunk
#define KC    (CCH*KK)        // 288 K per chunk
#define KBC   (KC/KBLK)       // 9 k-blocks per chunk
#define NCHUNK (Cin/CCH)      // 4 chunks

// ---------------------------------------------------------------------------
// Kernel 1: offset conv (18 ch) + modulator conv (9 ch, 2*sigmoid) fused.
// One thread per (b, ch, h, w); consecutive threads walk w -> coalesced x reads,
// identical weight address per wave -> scalar broadcast.
// ---------------------------------------------------------------------------
__global__ __launch_bounds__(256) void offmod_kernel(
    const float* __restrict__ x,
    const float* __restrict__ ow, const float* __restrict__ ob,
    const float* __restrict__ mw, const float* __restrict__ mb,
    float* __restrict__ offs, float* __restrict__ mask)
{
    int gid = blockIdx.x * 256 + threadIdx.x;
    const int total = Bn * 27 * Hh * Ww;
    if (gid >= total) return;
    int w  = gid % Ww;
    int h  = (gid / Ww) % Hh;
    int ch = (gid / (Ww * Hh)) % 27;
    int b  = gid / (Ww * Hh * 27);

    const float* wbase = (ch < 18) ? (ow + (size_t)ch * Cin * KK)
                                   : (mw + (size_t)(ch - 18) * Cin * KK);
    float acc = (ch < 18) ? ob[ch] : mb[ch - 18];
    const float* xb = x + (size_t)b * Cin * Hh * Ww;

    for (int c = 0; c < Cin; ++c) {
        const float* xcp = xb + (size_t)c * Hh * Ww;
        const float* wp  = wbase + c * KK;
#pragma unroll
        for (int ky = 0; ky < 3; ++ky) {
            int y = h - 1 + ky;
            if (y < 0 || y >= Hh) continue;
            const float* xr = xcp + (size_t)y * Ww;
#pragma unroll
            for (int kx = 0; kx < 3; ++kx) {
                int xx = w - 1 + kx;
                if (xx < 0 || xx >= Ww) continue;
                acc = fmaf(wp[ky * 3 + kx], xr[xx], acc);
            }
        }
    }
    if (ch < 18)
        offs[((size_t)(b * 18 + ch) * Hh + h) * Ww + w] = acc;
    else
        mask[((size_t)(b * 9 + (ch - 18)) * Hh + h) * Ww + w] =
            2.0f / (1.0f + __expf(-acc));
}

// K index inside a 16x32 f16 A/B fragment for (lane, half j) per CDNA5 ISA:
// VGPR v holds packed K pair; v0..3 -> K 0..7, v4..7 -> K 16..23 (lanes 0-15);
// lanes 16-31 add +8.
__device__ __forceinline__ int wmma_k_of(int lane, int j) {
    int v = j >> 1, lh = j & 1;
    int k = (v < 4) ? (2 * v + lh) : (16 + 2 * (v - 4) + lh);
    return k + ((lane & 16) ? 8 : 0);
}

// ---------------------------------------------------------------------------
// Kernel 2: pre-pack weight (256 x 1152 f32) into WMMA A-fragment layout, f16.
// fragA[((otile*NKB + kb)*32 + lane)*16 + j] = weight[otile*16 + lane%16][kb*32 + K(lane,j)]
// Main loop then loads one contiguous v16h (32B) per lane -> fully coalesced.
// ---------------------------------------------------------------------------
__global__ __launch_bounds__(256) void prepack_kernel(
    const float* __restrict__ wgt, _Float16* __restrict__ fragA)
{
    int gid = blockIdx.x * 256 + threadIdx.x;
    const int total = Oout * Ktot;          // 294912
    if (gid >= total) return;
    int j    = gid & 15;
    int lane = (gid >> 4) & 31;
    int kb   = (gid >> 9) % NKB;
    int ot   = gid / (512 * NKB);
    int M    = lane & 15;
    int K    = wmma_k_of(lane, j);
    int kg   = kb * KBLK + K;
    fragA[gid] = (_Float16)wgt[(size_t)(ot * 16 + M) * Ktot + kg];
}

// ---------------------------------------------------------------------------
// Kernel 3: bilinear-sample + 16x16x32 f16 WMMA contraction.
// Block = 16 pixels (one w-tile). 8 waves, each owning 32 Cout rows.
// ---------------------------------------------------------------------------
__global__ __launch_bounds__(256) void deform_wmma_kernel(
    const float* __restrict__ x, const _Float16* __restrict__ fragA,
    const float* __restrict__ offs, const float* __restrict__ mask,
    float* __restrict__ out)
{
    __shared__ int   sI[4][KK * NPIX];      // 4 bilinear corner indices
    __shared__ float sWt[4][KK * NPIX];     // 4 corner weights (mask folded in)
    __shared__ v16h  sB[2][KBC * 32];       // double-buffered B fragments

    const int tid    = threadIdx.x;
    const int tileId = blockIdx.x;
    const int wt = tileId % (Ww / NPIX);
    const int h  = (tileId / (Ww / NPIX)) % Hh;
    const int b  = tileId / ((Ww / NPIX) * Hh);
    const int w0 = wt * NPIX;

    // ---- Phase A: per-(tap,pixel) bilinear metadata (144 work items) ----
    if (tid < KK * NPIX) {
        int p  = tid & 15;
        int k  = tid >> 4;
        int wc = w0 + p;
        size_t pix = (size_t)h * Ww + wc;
        float oy = offs[((size_t)(b * 18) + 2 * k    ) * (Hh * Ww) + pix];
        float ox = offs[((size_t)(b * 18) + 2 * k + 1) * (Hh * Ww) + pix];
        float m  = mask[((size_t)(b * 9)  + k        ) * (Hh * Ww) + pix];
        float py = (float)(h  - 1 + k / 3) + oy;
        float px = (float)(wc - 1 + k % 3) + ox;
        float fy = floorf(py), fx = floorf(px);
        int y0 = (int)fy, x0 = (int)fx;
        float wy = py - fy, wx = px - fx;
        float cw[4] = { (1.f - wy) * (1.f - wx) * m, (1.f - wy) * wx * m,
                        wy * (1.f - wx) * m,         wy * wx * m };
#pragma unroll
        for (int cr = 0; cr < 4; ++cr) {
            int yy = y0 + (cr >> 1);
            int xx = x0 + (cr & 1);
            bool valid = (yy >= 0) && (yy < Hh) && (xx >= 0) && (xx < Ww);
            int yc = min(max(yy, 0), Hh - 1);
            int xc = min(max(xx, 0), Ww - 1);
            sI[cr][tid]  = yc * Ww + xc;
            sWt[cr][tid] = valid ? cw[cr] : 0.0f;
        }
    }
    __syncthreads();

    const int wave = tid >> 5;
    const int lane = tid & 31;
    const int row  = lane & 15;
    const float* xb = x + (size_t)b * Cin * Hh * Ww;
    _Float16* sBh = (_Float16*)sB;

    v8f acc0 = {}, acc1 = {};

    // Sample one chunk (32 channels x 9 taps = 288 K) directly into
    // B-fragment layout in LDS: element e = kbl*512 + lane*16 + j.
    auto fill = [&](int bufsel, int cb) {
        _Float16* dst = sBh + (size_t)bufsel * (KBC * 32 * 16);
        for (int e = tid; e < KBC * 32 * 16; e += 256) {
            int j   = e & 15;
            int ln  = (e >> 4) & 31;
            int kbl = e >> 9;
            int p   = ln & 15;
            int K   = wmma_k_of(ln, j);
            int kkg = kbl * KBLK + K;          // 0..287 within chunk
            int c   = cb + kkg / KK;
            int tap = kkg % KK;
            int pi  = tap * NPIX + p;
            const float* xp = xb + (size_t)c * (Hh * Ww);
            float v = fmaf(sWt[0][pi], xp[sI[0][pi]],
                      fmaf(sWt[1][pi], xp[sI[1][pi]],
                      fmaf(sWt[2][pi], xp[sI[2][pi]],
                           sWt[3][pi] * xp[sI[3][pi]])));
            dst[e] = (_Float16)v;
        }
    };

    fill(0, 0);
    __syncthreads();

    int buf = 0;
    for (int ch = 0; ch < NCHUNK; ++ch) {
        if (ch + 1 < NCHUNK) fill(buf ^ 1, (ch + 1) * CCH);  // prefetch next chunk

        const int ot0 = wave * 2, ot1 = wave * 2 + 1;        // two 16-row Cout tiles
#pragma unroll
        for (int kbl = 0; kbl < KBC; ++kbl) {
            int kbg = ch * KBC + kbl;
            v16h a0 = *(const v16h*)(fragA + ((size_t)(ot0 * NKB + kbg) * 32 + lane) * 16);
            v16h a1 = *(const v16h*)(fragA + ((size_t)(ot1 * NKB + kbg) * 32 + lane) * 16);
            v16h bf = sB[buf][kbl * 32 + lane];
            acc0 = __builtin_amdgcn_wmma_f32_16x16x32_f16(
                       false, a0, false, bf, (short)0, acc0, false, false);
            acc1 = __builtin_amdgcn_wmma_f32_16x16x32_f16(
                       false, a1, false, bf, (short)0, acc1, false, false);
        }
        __syncthreads();
        buf ^= 1;
    }

    // ---- Epilogue: C/D layout = VGPR r -> M=r (lanes 0-15) / M=r+8 (lanes 16-31)
    const int mhalf = (lane < 16) ? 0 : 8;
#pragma unroll
    for (int r = 0; r < 8; ++r) {
        int M = r + mhalf;
        size_t o0 = ((size_t)(b * Oout + wave * 32 + M)      * Hh + h) * Ww + w0 + row;
        size_t o1 = ((size_t)(b * Oout + wave * 32 + 16 + M) * Hh + h) * Ww + w0 + row;
        out[o0] = acc0[r];
        out[o1] = acc1[r];
    }
}

// ---------------------------------------------------------------------------
extern "C" void kernel_launch(void* const* d_in, const int* in_sizes, int n_in,
                              void* d_out, int out_size, void* d_ws, size_t ws_size,
                              hipStream_t stream) {
    (void)in_sizes; (void)n_in; (void)out_size; (void)ws_size;
    const float* x   = (const float*)d_in[0];
    const float* ow  = (const float*)d_in[1];
    const float* ob  = (const float*)d_in[2];
    const float* mw  = (const float*)d_in[3];
    const float* mb  = (const float*)d_in[4];
    const float* wgt = (const float*)d_in[5];
    float* out = (float*)d_out;

    // Workspace partition: offs (4*18*128*128 f32) | mask (4*9*128*128 f32) |
    // fragA (256*1152 f16)  -> ~7.7 MB total.
    float*    offs  = (float*)d_ws;
    float*    mask  = offs + (size_t)Bn * 18 * Hh * Ww;
    _Float16* fragA = (_Float16*)(mask + (size_t)Bn * 9 * Hh * Ww);

    {
        int total = Bn * 27 * Hh * Ww;
        offmod_kernel<<<(total + 255) / 256, 256, 0, stream>>>(
            x, ow, ob, mw, mb, offs, mask);
    }
    {
        int total = Oout * Ktot;
        prepack_kernel<<<(total + 255) / 256, 256, 0, stream>>>(wgt, fragA);
    }
    {
        int blocks = Bn * Hh * (Ww / NPIX);   // 4096
        deform_wmma_kernel<<<blocks, 256, 0, stream>>>(x, fragA, offs, mask, out);
    }
}
